// localPatchClassifierV2_53893249630333
// MI455X (gfx1250) — matrix operand order repaired and checked by reading, not compiled, not run on hardware
//
#include <hip/hip_runtime.h>

typedef __attribute__((ext_vector_type(16))) __bf16 v16bf;
typedef __attribute__((ext_vector_type(8)))  __bf16 v8bf;
typedef __attribute__((ext_vector_type(4)))  __bf16 v4bf;
typedef __attribute__((ext_vector_type(8)))  float  v8f;

#define DIM     640
#define KSTEPS  20          // 640 / 32
#define PN      121
#define NWQ     75
#define WAY     5
#define SUPN    605         // shot * patch_num
#define NSTEPS  38          // ceil(605 / 16)
#define BROW    648         // LDS row stride in halfs (640 + 8 pad -> 4-dword bank skew)

__device__ __forceinline__ void ins3(float v, float& t0, float& t1, float& t2) {
  if (v > t0)      { t2 = t1; t1 = t0; t0 = v; }
  else if (v > t1) { t2 = t1; t1 = v; }
  else if (v > t2) { t2 = v; }
}

__device__ __forceinline__ v16bf loadB(const __bf16* brow, int k) {
  v8bf lo = *(const v8bf*)(brow + 32 * k);
  v8bf hi = *(const v8bf*)(brow + 32 * k + 16);
  return __builtin_shufflevector(lo, hi, 0,1,2,3,4,5,6,7,8,9,10,11,12,13,14,15);
}

// ---- pre-pass: support f32 -> bf16 (same [b,w,s,d] layout) into workspace -----------
__global__ __launch_bounds__(256)
void cvt_sup_kernel(const float* __restrict__ in, __bf16* __restrict__ outp, int n4) {
  int i = blockIdx.x * 256 + threadIdx.x;
  if (i < n4) {
    float4 f = ((const float4*)in)[i];
    v4bf o;
    o[0] = (__bf16)f.x; o[1] = (__bf16)f.y; o[2] = (__bf16)f.z; o[3] = (__bf16)f.w;
    ((v4bf*)outp)[i] = o;
  }
}

__global__ __launch_bounds__(256)
void lpc_wmma_kernel(const float* __restrict__ query,
                     const __bf16* __restrict__ supb,   // bf16 support in workspace
                     float* __restrict__ out) {
  // double-buffered support tile (bf16): 2 x 16 x 648 halfs = 40.5 KB
  __shared__ __align__(16) __bf16 sB[2][16 * BROW];
  __shared__ float rowsum[128];

  const int bid = blockIdx.x;
  const int w   = bid % WAY;
  const int q   = (bid / WAY) % NWQ;
  const int b   = bid / (WAY * NWQ);

  const int tid  = threadIdx.x;
  const int wave = tid >> 5;
  const int lane = tid & 31;
  const int half = lane >> 4;   // 0: lanes 0-15, 1: lanes 16-31
  const int l16  = lane & 15;

  // ---------------- A tile: 16 query-patch rows for this wave, kept in VGPRs as bf16 ----
  // ISA 16-bit A layout (16x32): lanes 0-15 hold M=l16, K = {0..7, 16..23};
  //                              lanes 16-31 hold M=l16, K = {8..15, 24..31}.
  const int prow = wave * 16 + l16;                 // patch row this lane represents
  const int prc  = (prow < PN) ? prow : (PN - 1);   // clamp; masked out of the final sum
  const float* qbase = query + (((size_t)(b * NWQ + q) * PN + prc) * DIM);
  const int off = half * 8;

  v16bf a[KSTEPS];
#pragma unroll
  for (int k = 0; k < KSTEPS; ++k) {
    const float* p = qbase + 32 * k + off;
    float4 x0 = *(const float4*)(p);       // K = 32k + off + 0..3
    float4 x1 = *(const float4*)(p + 4);   // K = 32k + off + 4..7
    float4 y0 = *(const float4*)(p + 16);  // K = 32k + 16 + off + 0..3
    float4 y1 = *(const float4*)(p + 20);  // K = 32k + 16 + off + 4..7
    v16bf t;
    t[0]  = (__bf16)x0.x; t[1]  = (__bf16)x0.y; t[2]  = (__bf16)x0.z; t[3]  = (__bf16)x0.w;
    t[4]  = (__bf16)x1.x; t[5]  = (__bf16)x1.y; t[6]  = (__bf16)x1.z; t[7]  = (__bf16)x1.w;
    t[8]  = (__bf16)y0.x; t[9]  = (__bf16)y0.y; t[10] = (__bf16)y0.z; t[11] = (__bf16)y0.w;
    t[12] = (__bf16)y1.x; t[13] = (__bf16)y1.y; t[14] = (__bf16)y1.z; t[15] = (__bf16)y1.w;
    a[k] = t;
  }

  // Running top-3 per C-register slot j (row = j + 8*half for this lane, fixed for the kernel)
  float t0[8], t1[8], t2[8];
#pragma unroll
  for (int j = 0; j < 8; ++j) {
    t0[j] = -__builtin_inff(); t1[j] = -__builtin_inff(); t2[j] = -__builtin_inff();
  }

  const __bf16* sbase = supb + (size_t)(b * WAY + w) * (size_t)SUPN * DIM;

  // --- staging split: global->VGPR at iteration top, VGPR->LDS after compute ---------
  // per thread: 5 b128 chunks of the 16x640 bf16 tile (1280 chunks / 256 threads)
  v8bf stv[5];
  auto stageLoad = [&](int tile) {
    const int s0 = tile * 16;
#pragma unroll
    for (int i = 0; i < 5; ++i) {
      int idx = tid + i * 256;
      int e   = idx * 8;
      int row = e / DIM;
      int kk  = e - row * DIM;
      int s   = s0 + row;
      if (s >= SUPN) s = SUPN - 1;     // clamp; column masked at merge
      stv[i] = *(const v8bf*)(sbase + (size_t)s * DIM + kk);
    }
  };
  auto stageStore = [&](int buf) {
#pragma unroll
    for (int i = 0; i < 5; ++i) {
      int idx = tid + i * 256;
      int e   = idx * 8;
      int row = e / DIM;
      int kk  = e - row * DIM;
      *(v8bf*)(&sB[buf][row * BROW + kk]) = stv[i];
    }
  };

  stageLoad(0);
  stageStore(0);
  __syncthreads();

  for (int ns = 0; ns < NSTEPS; ++ns) {
    const bool pre = (ns + 1 < NSTEPS);
    if (pre) stageLoad(ns + 1);        // global loads in flight across the whole compute

    // --- GEMM micro-tile: C(16x16) += A(16x640) * B(640x16), 20 x wmma bf16.
    //     Cross-group software pipeline, enforced with sched_barrier(0):
    //     load group g+1 into the alternate buffer set, then run group g's WMMAs.
    // ISA 16-bit B layout (32x16): lanes 0-15 hold N=l16, K = 32k+{0..7, 16..23};
    //                              lanes 16-31 hold N=l16, K = 32k+{8..15, 24..31}.
    const __bf16* brow = &sB[ns & 1][l16 * BROW + 8 * half];
    v8f c0 = {0.f,0.f,0.f,0.f,0.f,0.f,0.f,0.f};
    v8f c1 = {0.f,0.f,0.f,0.f,0.f,0.f,0.f,0.f};
    v16bf B0[4], B1[4];

    auto LG = [&](v16bf* B, int kb) {
      B[0] = loadB(brow, kb);     B[1] = loadB(brow, kb + 1);
      B[2] = loadB(brow, kb + 2); B[3] = loadB(brow, kb + 3);
    };
    auto MG = [&](const v16bf* B, int kb) {
      c0 = __builtin_amdgcn_wmma_f32_16x16x32_bf16(false, a[kb],     false, B[0], (short)0, c0, false, false);
      c1 = __builtin_amdgcn_wmma_f32_16x16x32_bf16(false, a[kb + 1], false, B[1], (short)0, c1, false, false);
      c0 = __builtin_amdgcn_wmma_f32_16x16x32_bf16(false, a[kb + 2], false, B[2], (short)0, c0, false, false);
      c1 = __builtin_amdgcn_wmma_f32_16x16x32_bf16(false, a[kb + 3], false, B[3], (short)0, c1, false, false);
    };

    LG(B0, 0);
    __builtin_amdgcn_sched_barrier(0);
    LG(B1, 4);
    __builtin_amdgcn_sched_barrier(0);
    MG(B0, 0);
    LG(B0, 8);
    __builtin_amdgcn_sched_barrier(0);
    MG(B1, 4);
    LG(B1, 12);
    __builtin_amdgcn_sched_barrier(0);
    MG(B0, 8);
    LG(B0, 16);
    __builtin_amdgcn_sched_barrier(0);
    MG(B1, 12);
    MG(B0, 16);

    // --- fused top-3 epilogue: lane's value for slot j is (row j+8*half, col s0+l16)
    const bool colv = (ns * 16 + l16) < SUPN;
#pragma unroll
    for (int j = 0; j < 8; ++j) {
      float v = colv ? (c0[j] + c1[j]) : -__builtin_inff();
      ins3(v, t0[j], t1[j], t2[j]);
    }

    // publish next tile only now: its ds_stores never gate the compute loop's dscnt waits
    if (pre) stageStore((ns + 1) & 1);
    __syncthreads();  // staging of ns+1 visible; reads of buf[ns&1] done before re-staging
  }

  // --- merge top-3 across the 16 lanes of each half (butterfly, stays within half)
#pragma unroll
  for (int m = 8; m >= 1; m >>= 1) {
#pragma unroll
    for (int j = 0; j < 8; ++j) {
      float o0 = __shfl_xor(t0[j], m, 32);
      float o1 = __shfl_xor(t1[j], m, 32);
      float o2 = __shfl_xor(t2[j], m, 32);
      ins3(o0, t0[j], t1[j], t2[j]);
      ins3(o1, t0[j], t1[j], t2[j]);
      ins3(o2, t0[j], t1[j], t2[j]);
    }
  }

  // --- per-row top-3 sums to LDS (one writer per half)
  if (l16 == 0) {
#pragma unroll
    for (int j = 0; j < 8; ++j) {
      int r = wave * 16 + j + 8 * half;      // row within the 128-row block
      rowsum[r] = t0[j] + t1[j] + t2[j];
    }
  }
  __syncthreads();

  // --- wave 0: mean over 121 patches and k=3, single deterministic store
  if (wave == 0) {
    float s = 0.f;
#pragma unroll
    for (int i = 0; i < 4; ++i) {
      int p = lane + i * 32;
      if (p < PN) s += rowsum[p];
    }
#pragma unroll
    for (int m = 16; m >= 1; m >>= 1) s += __shfl_xor(s, m, 32);
    if (lane == 0)
      out[(b * NWQ + q) * WAY + w] = s * (1.0f / (PN * 3.0f));
  }
}

extern "C" void kernel_launch(void* const* d_in, const int* in_sizes, int n_in,
                              void* d_out, int out_size, void* d_ws, size_t ws_size,
                              hipStream_t stream) {
  (void)in_sizes; (void)n_in; (void)out_size; (void)ws_size;
  const float* query = (const float*)d_in[0];  // [8, 75, 121, 640] f32
  const float* sup   = (const float*)d_in[1];  // [8, 5, 5, 121, 640] f32
  float* out  = (float*)d_out;                 // [8, 75, 5] f32
  __bf16* supb = (__bf16*)d_ws;                // 15,488,000 bf16 = 31 MB workspace

  // pre-pass: support f32 -> bf16 (memory bound, ~93 MB traffic)
  const int n4 = (8 * WAY * SUPN * DIM) / 4;   // 3,872,000 float4 chunks
  cvt_sup_kernel<<<(n4 + 255) / 256, 256, 0, stream>>>(sup, supb, n4);

  dim3 grid(8 * NWQ * WAY);   // one block per (b, q, w)
  dim3 block(256);            // 8 waves; wave i owns query-patch rows [16i, 16i+16)
  lpc_wmma_kernel<<<grid, block, 0, stream>>>(query, supb, out);
}
